// FuelBurnPredictor_51840255262893
// MI455X (gfx1250) — compile-verified, matrix operand order checked by reading, not run on hardware
//
#include <hip/hip_runtime.h>
#include <hip/hip_bf16.h>

// ---------------- problem constants (from reference setup_inputs) -----------
#define T_TOK   32768
#define D_IN    16
#define AC_DIM  32
#define D_CAT   48
#define H_DIM   1024
#define NH      12
#define KD      768     // H * 0.75
#define VD      1536    // KD * 2
#define HK      64      // KD / NH
#define HV      128     // VD / NH
#define NSEG    16

typedef unsigned short u16;
typedef __attribute__((ext_vector_type(16))) __bf16 bf16x16;
typedef __attribute__((ext_vector_type(8)))  float  f32x8;
typedef __attribute__((ext_vector_type(8)))  u16    u16x8;
typedef __attribute__((ext_vector_type(4)))  unsigned int u32x4;
typedef __attribute__((ext_vector_type(8)))  int    i32x8;
typedef __attribute__((ext_vector_type(4)))  int    i32x4;

__device__ __forceinline__ u16 f2bf(float f) {
    unsigned int u = __float_as_uint(f);
    u += 0x7FFFu + ((u >> 16) & 1u);      // round-to-nearest-even
    return (u16)(u >> 16);
}
__device__ __forceinline__ float bf2f(u16 h) {
    return __uint_as_float(((unsigned int)h) << 16);
}
__device__ __forceinline__ float sigmoidf_(float x) { return 1.0f / (1.0f + expf(-x)); }

// CDNA5 async global->LDS copy (ASYNCcnt-tracked, no VGPR staging).
__device__ __forceinline__ void async_load_b128(unsigned lds_addr, const void* gptr) {
    asm volatile("global_load_async_to_lds_b128 %0, %1, off"
                 :: "v"(lds_addr), "v"((unsigned long long)(size_t)gptr)
                 : "memory");
}
#define S_WAIT_ASYNCCNT(n) asm volatile("s_wait_asynccnt " #n ::: "memory")

// ---------------- CDNA5 Tensor Data Mover (TDM) 2D tile load ----------------
#if __has_builtin(__builtin_amdgcn_tensor_load_to_lds)
#define HAVE_TDM 1
// Load a tile_h x tile_w tile of 2-byte elements from a row-major tensor with
// row stride row_stride (elements) into contiguous LDS at lds_addr.
__device__ __forceinline__ void tdm_load_2d(unsigned lds_addr, const void* gaddr,
                                            unsigned row_stride, unsigned tile_w,
                                            unsigned tile_h) {
    unsigned long long ga = (unsigned long long)(size_t)gaddr;
    u32x4 g0;
    g0.x = 1u;                                           // count=1, user descriptor
    g0.y = lds_addr;                                     // lds_addr [63:32]
    g0.z = (unsigned)ga;                                 // global_addr lo
    g0.w = (unsigned)((ga >> 32) & 0x01FFFFFFu) | (2u << 30);  // addr hi | type=2
    i32x8 g1;
    g1[0] = (int)(1u << 16);                             // data_size=1 (2 bytes)
    g1[1] = (int)((row_stride & 0xFFFFu) << 16);         // tensor_dim0[15:0] @bits63:48
    g1[2] = (int)((row_stride >> 16) |                   // tensor_dim0[31:16]
                  (((unsigned)T_TOK & 0xFFFFu) << 16));  // tensor_dim1[15:0]
    g1[3] = (int)(((unsigned)T_TOK >> 16) |              // tensor_dim1[31:16]
                  (tile_w << 16));                       // tile_dim0
    g1[4] = (int)(tile_h & 0xFFFFu);                     // tile_dim1 (tile_dim2=0)
    g1[5] = (int)row_stride;                             // tensor_dim0_stride lo
    g1[6] = 0;
    g1[7] = 0;
    i32x4 z4 = {0, 0, 0, 0};                             // groups 2/3: unused (2D)
#if defined(__clang_major__) && (__clang_major__ >= 23)
    i32x8 z8 = {0, 0, 0, 0, 0, 0, 0, 0};
    __builtin_amdgcn_tensor_load_to_lds(g0, g1, z4, z4, z8, 0);
#else
    __builtin_amdgcn_tensor_load_to_lds(g0, g1, z4, z4, 0);
#endif
}
#else
#define HAVE_TDM 0
#endif

// ---------------- kernel 0: segment bookkeeping -----------------------------
__global__ __launch_bounds__(256) void seg_kernel(const int* __restrict__ cu,
                                                  int* __restrict__ seg,
                                                  int* __restrict__ endslot) {
    int t = blockIdx.x * 256 + threadIdx.x;
    int s = 0;
    #pragma unroll
    for (int i = 1; i <= NSEG; ++i)
        if (t >= cu[i]) s = i;            // t < cu[NSEG] == T, so s <= NSEG-1
    seg[t]     = s;
    endslot[t] = (t == cu[s + 1] - 1) ? s : -1;
}

// ---------------- kernel 1: weight f32 -> bf16 with transpose ---------------
__global__ __launch_bounds__(256) void wconvT_kernel(const float* __restrict__ W,
                                                     u16* __restrict__ WT,
                                                     int K, int N) {
    size_t i = (size_t)blockIdx.x * 256 + threadIdx.x;
    int k2 = (int)(i / N), n = (int)(i % N);
    WT[(size_t)n * K + k2] = f2bf(W[i]);
}

// ---------------- kernel 2: embed GEMM + RMSNorm ----------------------------
__global__ __launch_bounds__(256) void prologue_kernel(
    const float* __restrict__ x, const int* __restrict__ seg,
    const int* __restrict__ endslot, const int* __restrict__ acidx,
    const float* __restrict__ ac_table, const float* __restrict__ W_in,
    const float* __restrict__ b_in, const float* __restrict__ norm_w,
    u16* __restrict__ hn_bf, float* __restrict__ h_res) {
    const int t = blockIdx.x;
    const int tid = threadIdx.x;
    __shared__ float xin[D_CAT];
    __shared__ float red[256];
    if (tid < D_CAT) {
        if (tid < D_IN) xin[tid] = x[(size_t)t * D_IN + tid];
        else            xin[tid] = ac_table[(size_t)acidx[seg[t]] * AC_DIM + (tid - D_IN)];
    }
    __syncthreads();
    float vals[4]; float ssq = 0.f;
    #pragma unroll
    for (int j = 0; j < 4; ++j) {
        int col = tid + 256 * j;
        float a = b_in[col];
        #pragma unroll
        for (int k2 = 0; k2 < D_CAT; ++k2) a += xin[k2] * W_in[(size_t)k2 * H_DIM + col];
        vals[j] = a; ssq += a * a;
    }
    red[tid] = ssq; __syncthreads();
    for (int st = 128; st > 0; st >>= 1) { if (tid < st) red[tid] += red[tid + st]; __syncthreads(); }
    float inv = rsqrtf(red[0] / (float)H_DIM + 1e-6f);
    int ps = endslot[t];
    #pragma unroll
    for (int j = 0; j < 4; ++j) {
        int col = tid + 256 * j;
        hn_bf[(size_t)t * H_DIM + col] = f2bf(vals[j] * inv * (1.0f + norm_w[col]));
        if (ps >= 0) h_res[(size_t)ps * H_DIM + col] = vals[j];   // pre-norm residual
    }
}

// ---------------- kernel 3: bf16 WMMA GEMM  C = A @ B^T ---------------------
// Block tile 128x128, 8 waves (4M x 2N), each wave 2x4 16x16 accumulators.
// Tiles staged via double-buffered GLOBAL_LOAD_ASYNC_TO_LDS_B128 (ASYNCcnt).
__device__ __forceinline__ bf16x16 load_frag(const u16* row, int lane) {
    const int kb = (lane >> 4) << 3;
    union { bf16x16 v; u16x8 h[2]; } u;
    u.h[0] = *(const u16x8*)(row + kb);
    u.h[1] = *(const u16x8*)(row + 16 + kb);
    return u.v;
}

__global__ __launch_bounds__(256) void gemm_bf16_wmma(
    const u16* __restrict__ A, const u16* __restrict__ BT,
    u16* __restrict__ C, int M, int N, int K) {
    __shared__ alignas(16) u16 As[2][128][32];
    __shared__ alignas(16) u16 Bs[2][128][32];
    const int tid  = threadIdx.x;
    const int lane = tid & 31;
    const int w    = tid >> 5;
    const int wm   = (w & 3) << 5;        // 0,32,64,96
    const int wn   = (w >> 2) << 6;       // 0,64
    const int m0 = blockIdx.y * 128;
    const int n0 = blockIdx.x * 128;
    const int lr = tid >> 1;              // 0..127
    const int lc = (tid & 1) << 4;        // 0 or 16

    const u16* gA = A  + (size_t)(m0 + lr) * K + lc;
    const u16* gB = BT + (size_t)(n0 + lr) * K + lc;

    unsigned la[2], lb[2];
    la[0] = (unsigned)(size_t)&As[0][lr][lc];
    la[1] = (unsigned)(size_t)&As[1][lr][lc];
    lb[0] = (unsigned)(size_t)&Bs[0][lr][lc];
    lb[1] = (unsigned)(size_t)&Bs[1][lr][lc];

    auto issue = [&](int k0, int buf) {
        async_load_b128(la[buf],      gA + k0);
        async_load_b128(la[buf] + 16, gA + k0 + 8);
        async_load_b128(lb[buf],      gB + k0);
        async_load_b128(lb[buf] + 16, gB + k0 + 8);
    };

    f32x8 acc[2][4] = {};
    auto compute = [&](int cur) {
        bf16x16 af[2], bfr[4];
        #pragma unroll
        for (int mi = 0; mi < 2; ++mi)
            af[mi]  = load_frag(&As[cur][wm + mi * 16 + (lane & 15)][0], lane);
        #pragma unroll
        for (int ni = 0; ni < 4; ++ni)
            bfr[ni] = load_frag(&Bs[cur][wn + ni * 16 + (lane & 15)][0], lane);
        #pragma unroll
        for (int mi = 0; mi < 2; ++mi)
            #pragma unroll
            for (int ni = 0; ni < 4; ++ni)
                acc[mi][ni] = __builtin_amdgcn_wmma_f32_16x16x32_bf16(
                    false, af[mi], false, bfr[ni], (short)0, acc[mi][ni], false, false);
    };

    issue(0, 0);
    for (int k0 = 0; k0 + 32 < K; k0 += 32) {           // steady state, branch-free
        const int cur = (k0 >> 5) & 1;
        issue(k0 + 32, cur ^ 1);
        S_WAIT_ASYNCCNT(4);       // current buffer landed; next 4 stay in flight
        __syncthreads();
        compute(cur);
        __syncthreads();          // reads done before buffer is overwritten
    }
    S_WAIT_ASYNCCNT(0);           // final tile
    __syncthreads();
    compute(((K - 32) >> 5) & 1);

    #pragma unroll
    for (int mi = 0; mi < 2; ++mi)
        #pragma unroll
        for (int ni = 0; ni < 4; ++ni)
            #pragma unroll
            for (int i = 0; i < 8; ++i) {
                int row = m0 + wm + mi * 16 + i + ((lane >> 4) << 3);
                int col = n0 + wn + ni * 16 + (lane & 15);
                C[(size_t)row * N + col] = f2bf(acc[mi][ni][i]);
            }
}

// ---------------- kernel 4: short conv + SiLU (+ per-head l2norm) -----------
__global__ __launch_bounds__(256) void conv_qk_kernel(
    const u16* __restrict__ raw, const float* __restrict__ cw,
    const int* __restrict__ seg, u16* __restrict__ outp, float scale) {
    const int t = blockIdx.x;
    const int tid = threadIdx.x;
    __shared__ float hsum[NH];
    if (tid < NH) hsum[tid] = 0.f;
    __syncthreads();
    const int st0 = seg[t];
    float vals[3];
    #pragma unroll
    for (int j = 0; j < 3; ++j) {
        int c = tid + 256 * j;
        float a = 0.f;
        #pragma unroll
        for (int s = 0; s < 4; ++s) {
            int ts = t - s;
            if (ts >= 0 && seg[ts] == st0)
                a += bf2f(raw[(size_t)ts * KD + c]) * cw[c * 4 + (3 - s)];
        }
        a = a * sigmoidf_(a);                     // silu
        vals[j] = a;
        atomicAdd(&hsum[c >> 6], a * a);          // ds_add_f32 per-head sumsq
    }
    __syncthreads();
    #pragma unroll
    for (int j = 0; j < 3; ++j) {
        int c = tid + 256 * j;
        float inv = rsqrtf(hsum[c >> 6] + 1e-6f);
        outp[(size_t)t * KD + c] = f2bf(vals[j] * inv * scale);
    }
}

__global__ __launch_bounds__(256) void conv_v_kernel(
    const u16* __restrict__ raw, const float* __restrict__ cw,
    const int* __restrict__ seg, u16* __restrict__ outp) {
    size_t idx = (size_t)blockIdx.x * 256 + threadIdx.x;
    int t = (int)(idx / VD), c = (int)(idx % VD);
    int st0 = seg[t];
    float a = 0.f;
    #pragma unroll
    for (int s = 0; s < 4; ++s) {
        int ts = t - s;
        if (ts >= 0 && seg[ts] == st0)
            a += bf2f(raw[(size_t)ts * VD + c]) * cw[c * 4 + (3 - s)];
    }
    a = a * sigmoidf_(a);
    outp[idx] = f2bf(a);
}

// ---------------- kernel 5: beta & decay ------------------------------------
__global__ __launch_bounds__(256) void betag_kernel(
    const u16* __restrict__ hn, const float* __restrict__ Wb,
    const float* __restrict__ Wa, const float* __restrict__ A_log,
    const float* __restrict__ dt_bias,
    float* __restrict__ beta, float* __restrict__ decay) {
    const int t = blockIdx.x; const int tid = threadIdx.x;
    __shared__ float sB[NH], sA[NH];
    if (tid < NH) { sB[tid] = 0.f; sA[tid] = 0.f; }
    __syncthreads();
    float aB[NH] = {}; float aA[NH] = {};
    #pragma unroll
    for (int j = 0; j < 4; ++j) {
        int k2 = tid + 256 * j;
        float hv = bf2f(hn[(size_t)t * H_DIM + k2]);
        #pragma unroll
        for (int h = 0; h < NH; ++h) {
            aB[h] += hv * Wb[k2 * NH + h];
            aA[h] += hv * Wa[k2 * NH + h];
        }
    }
    #pragma unroll
    for (int h = 0; h < NH; ++h) { atomicAdd(&sB[h], aB[h]); atomicAdd(&sA[h], aA[h]); }
    __syncthreads();
    if (tid < NH) {
        float apre = sA[tid] + dt_bias[tid];
        float sp   = fmaxf(apre, 0.f) + log1pf(expf(-fabsf(apre)));  // softplus
        beta[(size_t)t * NH + tid]  = sigmoidf_(sB[tid]);
        decay[(size_t)t * NH + tid] = expf(-expf(A_log[tid]) * sp);
    }
}

// ---------------- kernel 6: delta-rule scan, parallel over (head, segment) --
// Segments are independent (state resets at cu boundaries): 12 x 16 = 192
// blocks, ~2048 sequential steps each. Chunk staging uses the Tensor Data
// Mover (one 2D-tile DMA per q/k/v per chunk, TENSORcnt-tracked).
#define CH 16
__global__ __launch_bounds__(128) void scan_kernel(
    const u16* __restrict__ q, const u16* __restrict__ k,
    const u16* __restrict__ v, const float* __restrict__ decay,
    const float* __restrict__ beta, const int* __restrict__ cu,
    float* __restrict__ o_pool) {
    const int h = blockIdx.x;      // head
    const int s = blockIdx.y;      // segment
    const int tid = threadIdx.x;   // v-column 0..127
    const int tbeg = cu[s], tend = cu[s + 1];
    __shared__ alignas(16) u16 qsb[CH][HK], ksb[CH][HK], vsb[CH][HV];
    __shared__ float dec[CH], bet[CH];
    float S[HK];
    #pragma unroll
    for (int i = 0; i < HK; ++i) S[i] = 0.f;   // fresh state per segment

#if HAVE_TDM
    const unsigned lq = (unsigned)(size_t)&qsb[0][0];
    const unsigned lk = (unsigned)(size_t)&ksb[0][0];
    const unsigned lv = (unsigned)(size_t)&vsb[0][0];
#endif

    for (int t0 = tbeg; t0 < tend; t0 += CH) {
        const int nstep = (tend - t0 < CH) ? (tend - t0) : CH;
        if (t0 + CH < tend)
            __builtin_prefetch(v + (size_t)(t0 + CH) * VD + h * HV + tid, 0, 3);
        __syncthreads();
#if HAVE_TDM
        const int wv = tid >> 5;               // waves 0/1/2 issue one DMA each
        if (wv == 0)
            tdm_load_2d(lq, q + (size_t)t0 * KD + h * HK, KD, HK, (unsigned)nstep);
        else if (wv == 1)
            tdm_load_2d(lk, k + (size_t)t0 * KD + h * HK, KD, HK, (unsigned)nstep);
        else if (wv == 2)
            tdm_load_2d(lv, v + (size_t)t0 * VD + h * HV, VD, HV, (unsigned)nstep);
        __builtin_amdgcn_s_wait_tensorcnt(0);
#else
        #pragma unroll
        for (int j = 0; j < 8; ++j) {
            int idx = tid + 128 * j;
            int stp = idx >> 6, kk = idx & 63;
            if (stp < nstep) {
                qsb[stp][kk] = q[(size_t)(t0 + stp) * KD + h * HK + kk];
                ksb[stp][kk] = k[(size_t)(t0 + stp) * KD + h * HK + kk];
            }
        }
        #pragma unroll
        for (int j = 0; j < 16; ++j) {
            int idx = tid + 128 * j;
            int stp = idx >> 7, vcc = idx & 127;
            if (stp < nstep)
                vsb[stp][vcc] = v[(size_t)(t0 + stp) * VD + h * HV + vcc];
        }
#endif
        if (tid < nstep) {
            int t = t0 + tid;
            dec[tid] = decay[(size_t)t * NH + h];
            bet[tid] = beta[(size_t)t * NH + h];
        }
        __syncthreads();

        for (int stp = 0; stp < nstep; ++stp) {
            float d = dec[stp];
            float kv = 0.f;
            #pragma unroll
            for (int i = 0; i < HK; ++i) { S[i] *= d; kv += bf2f(ksb[stp][i]) * S[i]; }
            float upd = bet[stp] * (bf2f(vsb[stp][tid]) - kv);
            float o = 0.f;
            #pragma unroll
            for (int i = 0; i < HK; ++i) {
                S[i] += bf2f(ksb[stp][i]) * upd;
                o += bf2f(qsb[stp][i]) * S[i];
            }
            if (t0 + stp == tend - 1)
                o_pool[(size_t)s * VD + h * HV + tid] = o;   // pooled row only
        }
    }
}

// ---------------- kernel 7: 16-row epilogue ---------------------------------
__global__ __launch_bounds__(256) void epilogue_kernel(
    const float* __restrict__ o_pool, const float* __restrict__ o_norm_w,
    const u16* __restrict__ hn_bf, const float* __restrict__ Wg,
    const float* __restrict__ Wo, const float* __restrict__ h_res,
    const float* __restrict__ W_head, const float* __restrict__ b_head,
    const int* __restrict__ cu, float* __restrict__ out) {
    const int s = blockIdx.x; const int tid = threadIdx.x;
    const int t = cu[s + 1] - 1;
    __shared__ float hs[NH];
    __shared__ float og[VD];
    __shared__ float hnr[H_DIM];
    __shared__ float red[256];
    if (tid < NH) hs[tid] = 0.f;
    #pragma unroll
    for (int j = 0; j < 4; ++j) {
        int k2 = tid + 256 * j;
        hnr[k2] = bf2f(hn_bf[(size_t)t * H_DIM + k2]);
    }
    __syncthreads();
    float ov[6];
    #pragma unroll
    for (int j = 0; j < 6; ++j) {
        int i = tid + 256 * j;
        ov[j] = o_pool[(size_t)s * VD + i];
        atomicAdd(&hs[i >> 7], ov[j] * ov[j]);
    }
    __syncthreads();
    #pragma unroll
    for (int j = 0; j < 6; ++j) {
        int i = tid + 256 * j;
        float on = ov[j] * rsqrtf(hs[i >> 7] / (float)HV + 1e-5f) * o_norm_w[i & (HV - 1)];
        float g = 0.f;
        for (int k2 = 0; k2 < H_DIM; ++k2) g += hnr[k2] * Wg[(size_t)k2 * VD + i];
        og[i] = on * g * sigmoidf_(g);
    }
    __syncthreads();
    float partial = 0.f;
    #pragma unroll
    for (int j = 0; j < 4; ++j) {
        int c = tid + 256 * j;
        float hv = h_res[(size_t)s * H_DIM + c];
        for (int i = 0; i < VD; ++i) hv += og[i] * Wo[(size_t)i * H_DIM + c];
        partial += hv * W_head[c];
    }
    red[tid] = partial; __syncthreads();
    for (int stp = 128; stp > 0; stp >>= 1) { if (tid < stp) red[tid] += red[tid + stp]; __syncthreads(); }
    if (tid == 0) out[s] = red[0] + b_head[0];
}

// ---------------- host launch ------------------------------------------------
extern "C" void kernel_launch(void* const* d_in, const int* in_sizes, int n_in,
                              void* d_out, int out_size, void* d_ws, size_t ws_size,
                              hipStream_t stream) {
    (void)in_sizes; (void)n_in; (void)out_size; (void)ws_size;
    const float* x        = (const float*)d_in[0];
    const int*   cu       = (const int*)  d_in[1];
    const int*   acidx    = (const int*)  d_in[2];
    const float* ac_table = (const float*)d_in[3];
    const float* W_in     = (const float*)d_in[4];
    const float* b_in     = (const float*)d_in[5];
    const float* norm_w   = (const float*)d_in[6];
    const float* Wq       = (const float*)d_in[7];
    const float* Wk       = (const float*)d_in[8];
    const float* Wv       = (const float*)d_in[9];
    const float* conv_qw  = (const float*)d_in[10];
    const float* conv_kw  = (const float*)d_in[11];
    const float* conv_vw  = (const float*)d_in[12];
    const float* Wb       = (const float*)d_in[13];
    const float* Wa       = (const float*)d_in[14];
    const float* A_log    = (const float*)d_in[15];
    const float* dt_bias  = (const float*)d_in[16];
    const float* Wg       = (const float*)d_in[17];
    const float* o_norm_w = (const float*)d_in[18];
    const float* Wo       = (const float*)d_in[19];
    const float* W_head   = (const float*)d_in[20];
    const float* b_head   = (const float*)d_in[21];

    char* base = (char*)d_ws;
    size_t off = 0;
    auto alloc = [&](size_t bytes) -> char* {
        char* p = base + off;
        off = (off + bytes + 255) & ~(size_t)255;
        return p;
    };
    int*   seg     = (int*)  alloc((size_t)T_TOK * 4);
    int*   endslot = (int*)  alloc((size_t)T_TOK * 4);
    u16*   hn_bf   = (u16*)  alloc((size_t)T_TOK * H_DIM * 2);
    float* h_res   = (float*)alloc((size_t)NSEG * H_DIM * 4);
    u16*   WqT     = (u16*)  alloc((size_t)KD * H_DIM * 2);
    u16*   WkT     = (u16*)  alloc((size_t)KD * H_DIM * 2);
    u16*   WvT     = (u16*)  alloc((size_t)VD * H_DIM * 2);
    u16*   qraw    = (u16*)  alloc((size_t)T_TOK * KD * 2);
    u16*   kraw    = (u16*)  alloc((size_t)T_TOK * KD * 2);
    u16*   vraw    = (u16*)  alloc((size_t)T_TOK * VD * 2);
    u16*   qc      = (u16*)  alloc((size_t)T_TOK * KD * 2);
    u16*   kc      = (u16*)  alloc((size_t)T_TOK * KD * 2);
    u16*   vc      = (u16*)  alloc((size_t)T_TOK * VD * 2);
    float* betab   = (float*)alloc((size_t)T_TOK * NH * 4);
    float* decayb  = (float*)alloc((size_t)T_TOK * NH * 4);
    float* o_pool  = (float*)alloc((size_t)NSEG * VD * 4);

    seg_kernel<<<T_TOK / 256, 256, 0, stream>>>(cu, seg, endslot);
    wconvT_kernel<<<(H_DIM * KD) / 256, 256, 0, stream>>>(Wq, WqT, H_DIM, KD);
    wconvT_kernel<<<(H_DIM * KD) / 256, 256, 0, stream>>>(Wk, WkT, H_DIM, KD);
    wconvT_kernel<<<(H_DIM * VD) / 256, 256, 0, stream>>>(Wv, WvT, H_DIM, VD);
    prologue_kernel<<<T_TOK, 256, 0, stream>>>(x, seg, endslot, acidx, ac_table,
                                               W_in, b_in, norm_w, hn_bf, h_res);
    gemm_bf16_wmma<<<dim3(KD / 128, T_TOK / 128), 256, 0, stream>>>(hn_bf, WqT, qraw, T_TOK, KD, H_DIM);
    gemm_bf16_wmma<<<dim3(KD / 128, T_TOK / 128), 256, 0, stream>>>(hn_bf, WkT, kraw, T_TOK, KD, H_DIM);
    gemm_bf16_wmma<<<dim3(VD / 128, T_TOK / 128), 256, 0, stream>>>(hn_bf, WvT, vraw, T_TOK, VD, H_DIM);
    conv_qk_kernel<<<T_TOK, 256, 0, stream>>>(qraw, conv_qw, seg, qc, 0.125f /* HK^-0.5 */);
    conv_qk_kernel<<<T_TOK, 256, 0, stream>>>(kraw, conv_kw, seg, kc, 1.0f);
    conv_v_kernel<<<(T_TOK * VD) / 256, 256, 0, stream>>>(vraw, conv_vw, seg, vc);
    betag_kernel<<<T_TOK, 256, 0, stream>>>(hn_bf, Wb, Wa, A_log, dt_bias, betab, decayb);
    scan_kernel<<<dim3(NH, NSEG), 128, 0, stream>>>(qc, kc, vc, decayb, betab, cu, o_pool);
    epilogue_kernel<<<NSEG, 256, 0, stream>>>(o_pool, o_norm_w, hn_bf, Wg, Wo, h_res,
                                              W_head, b_head, cu, (float*)d_out);
}
